// Gemma4MoE_53601191854593
// MI455X (gfx1250) — matrix operand order repaired and checked by reading, not compiled, not run on hardware
//
#include <hip/hip_runtime.h>
#include <hip/hip_bf16.h>
#include <math.h>

// ---------------- CDNA5 WMMA types ----------------
typedef __attribute__((ext_vector_type(16))) __bf16 v16bf;
typedef __attribute__((ext_vector_type(8)))  __bf16 v8bf;
typedef __attribute__((ext_vector_type(8)))  float  v8f;
typedef __attribute__((ext_vector_type(4)))  int    v4i;

typedef __attribute__((address_space(1))) v4i as1_v4i;  // global int4*
typedef __attribute__((address_space(3))) v4i as3_v4i;  // LDS int4*

static constexpr int T = 8192, H = 2048, D = 2048, E = 8;
static constexpr int BM = 128, BN = 128, BK = 32;
static constexpr int LDSK = 40;           // bf16 elems per LDS row (80B, 16B-aligned)
static constexpr int TILE = BM * LDSK;    // elems per LDS tile buffer

// ---- async global->LDS copy of 16B per lane (CDNA5 GLOBAL_LOAD_ASYNC_TO_LDS_B128)
__device__ __forceinline__ void async_b128(const __bf16* gsrc, __bf16* ldst) {
#if __has_builtin(__builtin_amdgcn_global_load_async_to_lds_b128)
    __builtin_amdgcn_global_load_async_to_lds_b128(
        (as1_v4i*)gsrc, (as3_v4i*)ldst, 0, 0);
#else
    *(v8bf*)ldst = *(const v8bf*)gsrc;
#endif
}
__device__ __forceinline__ void wait_async() {
#if __has_builtin(__builtin_amdgcn_s_wait_asynccnt)
    __builtin_amdgcn_s_wait_asynccnt(0);
#endif
}

__device__ __forceinline__ v16bf make_frag(v8bf lo, v8bf hi) {
    v16bf r;
#pragma unroll
    for (int i = 0; i < 8; ++i) { r[i] = lo[i]; r[i + 8] = hi[i]; }
    return r;
}

__device__ __forceinline__ v8f wmma_bf16(v16bf a, v16bf b, v8f c) {
    return __builtin_amdgcn_wmma_f32_16x16x32_bf16(
        false, a, false, b, (short)0, c, false, false);
}

// A fragment (16x32 bf16): lanes 0-15 row M=l hold K{0..7,16..23};
// lanes 16-31 row M=l-16 hold K{8..15,24..31}.
__device__ __forceinline__ v16bf load_a_frag(const __bf16* tile, int row0, int lane) {
    const int row = row0 + (lane & 15);
    const int kb  = (lane < 16) ? 0 : 8;
    v8bf lo = *(const v8bf*)(tile + row * LDSK + kb);
    v8bf hi = *(const v8bf*)(tile + row * LDSK + kb + 16);
    return make_frag(lo, hi);
}

// B fragment (32x16 bf16): lanes 0-15 col N=l hold K{0..7,16..23};
// lanes 16-31 col N=l-16 hold K{8..15,24..31}.  LDS tile stored as [n][k].
__device__ __forceinline__ v16bf load_b_frag(const __bf16* tile, int col0, int lane) {
    const int col = col0 + (lane & 15);
    const int kb  = (lane < 16) ? 0 : 8;
    v8bf lo = *(const v8bf*)(tile + col * LDSK + kb);
    v8bf hi = *(const v8bf*)(tile + col * LDSK + kb + 16);
    return make_frag(lo, hi);
}

// pack 4 staged float4 (16 fp32) -> two v8bf LDS stores
__device__ __forceinline__ void cvt_store(const float4* st, __bf16* dst) {
#pragma unroll
    for (int h = 0; h < 2; ++h) {
        float4 f0 = st[2 * h + 0];
        float4 f1 = st[2 * h + 1];
        v8bf pk;
        pk[0] = (__bf16)f0.x; pk[1] = (__bf16)f0.y; pk[2] = (__bf16)f0.z; pk[3] = (__bf16)f0.w;
        pk[4] = (__bf16)f1.x; pk[5] = (__bf16)f1.y; pk[6] = (__bf16)f1.z; pk[7] = (__bf16)f1.w;
        *(v8bf*)(dst + 8 * h) = pk;
    }
}

// fast tanh-GELU: gelu_tanh(g) = g * sigmoid(1.5957691f * (g + 0.044715f g^3))
__device__ __forceinline__ float gelu_tanh(float g) {
    const float z2 = 1.5957691216f * (g + 0.044715f * g * g * g);
    return g * __builtin_amdgcn_rcpf(1.f + __expf(-z2));
}

// ---------------- Kernel 1: RMS-norm + router + top-2 + x->bf16 ----------------
__global__ __launch_bounds__(256) void moe_router_kernel(
    const float* __restrict__ x, const float* __restrict__ router_scale,
    const float* __restrict__ per_expert_scale, const float* __restrict__ router_w,
    __bf16* __restrict__ x_bf16, float* __restrict__ combine)
{
    const int t   = blockIdx.x;
    const int tid = threadIdx.x;
    __shared__ float red[256];
    __shared__ float lsh[E];

    const float* xr = x + (size_t)t * H;
    float xv[8];
    float ssq = 0.f;
#pragma unroll
    for (int i = 0; i < 8; ++i) {
        const int k = tid + i * 256;
        const float v = xr[k];
        xv[i] = v;
        ssq += v * v;
        x_bf16[(size_t)t * H + k] = (__bf16)v;
    }
    red[tid] = ssq;
    __syncthreads();
    for (int s = 128; s > 0; s >>= 1) {
        if (tid < s) red[tid] += red[tid + s];
        __syncthreads();
    }
    const float rinv = rsqrtf(red[0] / (float)H + 1e-6f) * rsqrtf((float)H);
    __syncthreads();

    float acc[E];
#pragma unroll
    for (int e = 0; e < E; ++e) acc[e] = 0.f;
#pragma unroll
    for (int i = 0; i < 8; ++i) {
        const int k = tid + i * 256;
        const float r = xv[i] * rinv * router_scale[k];
#pragma unroll
        for (int e = 0; e < E; ++e) acc[e] += r * router_w[e * H + k];
    }
#pragma unroll
    for (int e = 0; e < E; ++e) {
        red[tid] = acc[e];
        __syncthreads();
        for (int s = 128; s > 0; s >>= 1) {
            if (tid < s) red[tid] += red[tid + s];
            __syncthreads();
        }
        if (tid == 0) lsh[e] = red[0];
        __syncthreads();
    }

    if (tid == 0) {
        float m = lsh[0];
#pragma unroll
        for (int e = 1; e < E; ++e) m = fmaxf(m, lsh[e]);
        float p[E], sum = 0.f;
#pragma unroll
        for (int e = 0; e < E; ++e) { p[e] = __expf(lsh[e] - m); sum += p[e]; }
        int i0 = 0;
#pragma unroll
        for (int e = 1; e < E; ++e) if (lsh[e] > lsh[i0]) i0 = e;
        int i1 = (i0 == 0) ? 1 : 0;
#pragma unroll
        for (int e = 0; e < E; ++e) if (e != i0 && lsh[e] > lsh[i1]) i1 = e;
        float w0 = p[i0] / sum, w1 = p[i1] / sum;
        const float s01 = w0 + w1;
        w0 = (w0 / s01) * per_expert_scale[i0];
        w1 = (w1 / s01) * per_expert_scale[i1];
        float row[E];
#pragma unroll
        for (int e = 0; e < E; ++e) row[e] = 0.f;
        row[i0] = w0; row[i1] = w1;
#pragma unroll
        for (int e = 0; e < E; ++e) combine[(size_t)t * E + e] = row[e];
    }
}

// ---------------- Kernel 2: gate/up GEMM + GELU*up -> h (bf16) ----------------
__global__ __launch_bounds__(256) void moe_gateup_kernel(
    const __bf16* __restrict__ xb,   // [T][H] bf16
    const float*  __restrict__ w,    // gate_up_proj[e]: [2D][H] fp32
    __bf16* __restrict__ hbuf)       // [T][D] bf16
{
    __shared__ __bf16 As[2 * TILE];
    __shared__ __bf16 Bg[2 * TILE];
    __shared__ __bf16 Bu[2 * TILE];

    const int tid  = threadIdx.x;
    const int lane = tid & 31;
    const int wave = tid >> 5;
    const int wm   = wave & 3;   // 4 waves along M (32 rows each)
    const int wn   = wave >> 2;  // 2 waves along N (64 cols each)
    const int t0   = blockIdx.x * BM;
    const int n0   = blockIdx.y * BN;

    v8f accg[2][4], accu[2][4];
#pragma unroll
    for (int i = 0; i < 2; ++i)
#pragma unroll
        for (int j = 0; j < 4; ++j) { accg[i][j] = (v8f)0.f; accu[i][j] = (v8f)0.f; }

    const int lrow = tid >> 1;            // 0..127
    const int lkk  = (tid & 1) * 16;      // 0 or 16
    const int KT   = H / BK;

    float4 bg_st[4], bu_st[4];

    auto issue_a = [&](int kt, int buf) {
        const __bf16* src = xb + (size_t)(t0 + lrow) * H + kt * BK + lkk;
        __bf16* dst = As + buf * TILE + lrow * LDSK + lkk;
        async_b128(src, dst);
        async_b128(src + 8, dst + 8);
    };
    auto issue_b = [&](int kt) {
        const float* gs = w + (size_t)(n0 + lrow) * H + kt * BK + lkk;
        const float* us = w + (size_t)(D + n0 + lrow) * H + kt * BK + lkk;
#pragma unroll
        for (int v = 0; v < 4; ++v) {
            bg_st[v] = ((const float4*)gs)[v];
            bu_st[v] = ((const float4*)us)[v];
        }
    };
    auto store_b = [&](int buf) {
        cvt_store(bg_st, Bg + buf * TILE + lrow * LDSK + lkk);
        cvt_store(bu_st, Bu + buf * TILE + lrow * LDSK + lkk);
    };

    // prologue: fill buffer 0
    issue_a(0, 0);
    issue_b(0);
    store_b(0);
    wait_async();
    __syncthreads();

    int buf = 0;
    for (int kt = 0; kt < KT; ++kt) {
        const int nbuf = buf ^ 1;
        if (kt + 1 < KT) {
            issue_a(kt + 1, nbuf);   // async DMA overlaps WMMA below
            issue_b(kt + 1);         // global loads in flight during compute
        }

        const __bf16* a_t = As + buf * TILE;
        v16bf a0 = load_a_frag(a_t, wm * 32,      lane);
        v16bf a1 = load_a_frag(a_t, wm * 32 + 16, lane);
#pragma unroll
        for (int j = 0; j < 4; ++j) {
            const int c0 = wn * 64 + j * 16;
            v16bf bg = load_b_frag(Bg + buf * TILE, c0, lane);
            accg[0][j] = wmma_bf16(a0, bg, accg[0][j]);
            accg[1][j] = wmma_bf16(a1, bg, accg[1][j]);
            v16bf bu = load_b_frag(Bu + buf * TILE, c0, lane);
            accu[0][j] = wmma_bf16(a0, bu, accu[0][j]);
            accu[1][j] = wmma_bf16(a1, bu, accu[1][j]);
        }

        if (kt + 1 < KT) {
            store_b(nbuf);
            wait_async();
        }
        __syncthreads();
        buf = nbuf;
    }

    // Epilogue: h = gelu_tanh(gate) * up, store bf16
    const int rbase = (lane < 16) ? 0 : 8;
    const int cl    = lane & 15;
#pragma unroll
    for (int i = 0; i < 2; ++i) {
#pragma unroll
        for (int j = 0; j < 4; ++j) {
#pragma unroll
            for (int r = 0; r < 8; ++r) {
                const int row = t0 + wm * 32 + i * 16 + r + rbase;
                const int col = n0 + wn * 64 + j * 16 + cl;
                const float g = accg[i][j][r];
                const float u = accu[i][j][r];
                hbuf[(size_t)row * D + col] = (__bf16)(gelu_tanh(g) * u);
            }
        }
    }
}

// ---------------- Kernel 3: down GEMM, scale by combine, accumulate out ----------------
__global__ __launch_bounds__(256) void moe_down_kernel(
    const __bf16* __restrict__ hbuf,  // [T][D] bf16
    const float*  __restrict__ w,     // down_proj[e]: [H][D] fp32
    const float*  __restrict__ combine, int e,
    float* __restrict__ out, int accumulate)
{
    __shared__ __bf16 As[2 * TILE];
    __shared__ __bf16 Bs[2 * TILE];

    const int tid  = threadIdx.x;
    const int lane = tid & 31;
    const int wave = tid >> 5;
    const int wm   = wave & 3;
    const int wn   = wave >> 2;
    const int t0   = blockIdx.x * BM;
    const int n0   = blockIdx.y * BN;   // over H outputs

    v8f acc[2][4];
#pragma unroll
    for (int i = 0; i < 2; ++i)
#pragma unroll
        for (int j = 0; j < 4; ++j) acc[i][j] = (v8f)0.f;

    const int lrow = tid >> 1;
    const int lkk  = (tid & 1) * 16;
    const int KT   = D / BK;

    float4 b_st[4];

    auto issue_a = [&](int kt, int buf) {
        const __bf16* src = hbuf + (size_t)(t0 + lrow) * D + kt * BK + lkk;
        __bf16* dst = As + buf * TILE + lrow * LDSK + lkk;
        async_b128(src, dst);
        async_b128(src + 8, dst + 8);
    };
    auto issue_b = [&](int kt) {
        const float* gs = w + (size_t)(n0 + lrow) * D + kt * BK + lkk;
#pragma unroll
        for (int v = 0; v < 4; ++v) b_st[v] = ((const float4*)gs)[v];
    };
    auto store_b = [&](int buf) {
        cvt_store(b_st, Bs + buf * TILE + lrow * LDSK + lkk);
    };

    issue_a(0, 0);
    issue_b(0);
    store_b(0);
    wait_async();
    __syncthreads();

    int buf = 0;
    for (int kt = 0; kt < KT; ++kt) {
        const int nbuf = buf ^ 1;
        if (kt + 1 < KT) {
            issue_a(kt + 1, nbuf);
            issue_b(kt + 1);
        }

        const __bf16* a_t = As + buf * TILE;
        v16bf a0 = load_a_frag(a_t, wm * 32,      lane);
        v16bf a1 = load_a_frag(a_t, wm * 32 + 16, lane);
#pragma unroll
        for (int j = 0; j < 4; ++j) {
            v16bf b = load_b_frag(Bs + buf * TILE, wn * 64 + j * 16, lane);
            acc[0][j] = wmma_bf16(a0, b, acc[0][j]);
            acc[1][j] = wmma_bf16(a1, b, acc[1][j]);
        }

        if (kt + 1 < KT) {
            store_b(nbuf);
            wait_async();
        }
        __syncthreads();
        buf = nbuf;
    }

    const int rbase = (lane < 16) ? 0 : 8;
    const int cl    = lane & 15;
    float wt[2][8];
#pragma unroll
    for (int i = 0; i < 2; ++i)
#pragma unroll
        for (int r = 0; r < 8; ++r)
            wt[i][r] = combine[(size_t)(t0 + wm * 32 + i * 16 + r + rbase) * E + e];

#pragma unroll
    for (int i = 0; i < 2; ++i) {
#pragma unroll
        for (int j = 0; j < 4; ++j) {
#pragma unroll
            for (int r = 0; r < 8; ++r) {
                const int row = t0 + wm * 32 + i * 16 + r + rbase;
                const int col = n0 + wn * 64 + j * 16 + cl;
                const size_t idx = (size_t)row * H + col;
                const float v = wt[i][r] * acc[i][j][r];
                out[idx] = accumulate ? (out[idx] + v) : v;
            }
        }
    }
}

// ---------------- Host launcher ----------------
extern "C" void kernel_launch(void* const* d_in, const int* in_sizes, int n_in,
                              void* d_out, int out_size, void* d_ws, size_t ws_size,
                              hipStream_t stream) {
    (void)in_sizes; (void)n_in; (void)out_size; (void)ws_size;
    const float* x   = (const float*)d_in[0];
    const float* rsc = (const float*)d_in[1];
    const float* pes = (const float*)d_in[2];
    const float* rw  = (const float*)d_in[3];
    const float* gup = (const float*)d_in[4];
    const float* dwn = (const float*)d_in[5];
    float* out = (float*)d_out;

    char* ws = (char*)d_ws;
    __bf16* xb      = (__bf16*)ws;                                          // T*H*2
    __bf16* hb      = (__bf16*)(ws + (size_t)T * H * 2);                    // T*D*2
    float*  combine = (float*)(ws + (size_t)T * H * 2 + (size_t)T * D * 2); // T*E*4

    moe_router_kernel<<<T, 256, 0, stream>>>(x, rsc, pes, rw, xb, combine);

    for (int e = 0; e < E; ++e) {
        moe_gateup_kernel<<<dim3(T / BM, D / BN), 256, 0, stream>>>(
            xb, gup + (size_t)e * 2 * D * H, hb);
        moe_down_kernel<<<dim3(T / BM, H / BN), 256, 0, stream>>>(
            hb, dwn + (size_t)e * (size_t)H * D, combine, e, out, e != 0);
    }
}
